// LoKrLayer_893353197813
// MI455X (gfx1250) — compile-verified, matrix operand order checked by reading, not run on hardware
//
#include <hip/hip_runtime.h>

// LoKr: out = (x @ kron(w1, w2)^T) * 0.25
// Factorized: Y = X'(131072x512) @ w2^T (bf16 WMMA, f32 acc), then per-token
// 8x8 w1 combine done entirely in-lane across the 8 C-fragment VGPRs.
// B (w2) is pre-converted to bf16 in d_ws and DMA'd to LDS with
// global_load_async_to_lds_b128 (ASYNCcnt); A (x) converts through VGPRs.

typedef __attribute__((ext_vector_type(16))) __bf16 v16bf;
typedef __attribute__((ext_vector_type(8)))  float  v8f;
typedef __attribute__((ext_vector_type(2)))  __bf16 v2bf;

#define BM 128
#define BN 128
#define BK 32
#define KDIM 512
#define NDIM 512
#define ROWS 131072          // 4*4096*8 rows of X'
#define LSTRIDE 20           // uints per LDS row = 80B: 16B aligned, non-pow2
#define NSTEP (KDIM / BK)    // 16 k-steps

__device__ __forceinline__ unsigned pack_bf16x2(float a, float b) {
#if __has_builtin(__builtin_amdgcn_cvt_pk_bf16_f32)
  v2bf p = __builtin_amdgcn_cvt_pk_bf16_f32(a, b);
  return __builtin_bit_cast(unsigned, p);
#else
  unsigned r;
  asm("v_cvt_pk_bf16_f32 %0, %1, %2" : "=v"(r) : "v"(a), "v"(b));
  return r;
#endif
}

// Async global -> LDS DMA (per-lane 16B), saddr + 32-bit voffset form.
__device__ __forceinline__ void async_load_b128(unsigned lds_off,
                                                const void* sbase,
                                                unsigned goff) {
  asm volatile("global_load_async_to_lds_b128 %0, %1, %2"
               :: "v"(lds_off), "v"(goff), "s"(sbase)
               : "memory");
}
__device__ __forceinline__ void wait_async0() {
  asm volatile("s_wait_asynccnt 0x0" ::: "memory");
}

union FragBF {
  uint4 q[2];
  v16bf v;
};

template <bool PRECONV>
__global__ __launch_bounds__(256)
void lokr_wmma_kernel(const float* __restrict__ x,
                      const float* __restrict__ w1,
                      const float* __restrict__ w2,
                      const unsigned short* __restrict__ w2bf,
                      float* __restrict__ out) {
  // double-buffered bf16 panels: 2 * (10,240 + 10,240) = 40,960 B
  __shared__ unsigned Als[2][BM * LSTRIDE];
  __shared__ unsigned Bls[2][BN * LSTRIDE];

  const int t    = threadIdx.x;
  const int lane = t & 31;
  const int wid  = t >> 5;       // 8 waves
  const int wm   = wid & 1;      // 2 waves along M (64 rows each)
  const int wn   = wid >> 1;     // 4 waves along N (32 cols each)
  const int g    = lane >> 4;    // K-group select (ISA 16-bit A/B layout)
  const int lm   = lane & 15;    // M (resp. N) within 16-wide tile

  const int m0 = blockIdx.x * BM;   // row tile in X' space
  const int n0 = blockIdx.y * BN;   // q tile in 0..511

  // ---- per-thread coordinates, computed once ----
  unsigned aoff[4], boff[4], lsoff[4];
#pragma unroll
  for (int i = 0; i < 4; ++i) {
    const int f   = t + i * 256;   // 0..1023
    const int row = f >> 3;        // 0..127
    const int c4  = f & 7;         // float4 within 32-wide K slab
    aoff[i]  = (unsigned)((m0 + row) * KDIM + c4 * 4);
    boff[i]  = (unsigned)((n0 + row) * KDIM + c4 * 4);
    lsoff[i] = (unsigned)(row * LSTRIDE + c4 * 2);
  }
  // async B chunk coords (PRECONV): 512 x 16B chunks per panel, 2 per thread
  unsigned bgoff[2], bloff[2];
#pragma unroll
  for (int j = 0; j < 2; ++j) {
    const int ch  = t + j * 256;   // 0..511
    const int row = ch >> 2;       // 0..127
    const int cc  = ch & 3;        // 16B chunk within the 64B bf16 row slab
    bgoff[j] = (unsigned)((n0 + row) * (KDIM * 2) + cc * 16); // bytes in w2bf
    bloff[j] = (unsigned)(row * (LSTRIDE * 4) + cc * 16);     // bytes in panel
  }
  const unsigned blsbase0 = (unsigned)(size_t)&Bls[0][0];     // LDS byte offset
  const unsigned blsbase1 = (unsigned)(size_t)&Bls[1][0];

  unsigned afrag[4], bfrag[2];
#pragma unroll
  for (int ms = 0; ms < 4; ++ms)
    afrag[ms] = (unsigned)((wm * 64 + ms * 16 + lm) * LSTRIDE + g * 4);
#pragma unroll
  for (int ns = 0; ns < 2; ++ns)
    bfrag[ns] = (unsigned)((wn * 32 + ns * 16 + lm) * LSTRIDE + g * 4);

  const v8f zero = {0.f, 0.f, 0.f, 0.f, 0.f, 0.f, 0.f, 0.f};
  v8f c[4][2];
#pragma unroll
  for (int ms = 0; ms < 4; ++ms)
#pragma unroll
    for (int ns = 0; ns < 2; ++ns) c[ms][ns] = zero;

  // ---- prologue: panel 0 ----
  float4 ra[4], rb[4];
  if (PRECONV) {
#pragma unroll
    for (int j = 0; j < 2; ++j)
      async_load_b128(blsbase0 + bloff[j], w2bf, bgoff[j]);
  }
#pragma unroll
  for (int i = 0; i < 4; ++i) {
    ra[i] = *(const float4*)(x + aoff[i]);
    if (!PRECONV) rb[i] = *(const float4*)(w2 + boff[i]);
  }
#pragma unroll
  for (int i = 0; i < 4; ++i) {
    uint2 ap; ap.x = pack_bf16x2(ra[i].x, ra[i].y); ap.y = pack_bf16x2(ra[i].z, ra[i].w);
    *(uint2*)&Als[0][lsoff[i]] = ap;
    if (!PRECONV) {
      uint2 bp; bp.x = pack_bf16x2(rb[i].x, rb[i].y); bp.y = pack_bf16x2(rb[i].z, rb[i].w);
      *(uint2*)&Bls[0][lsoff[i]] = bp;
    }
  }
  if (PRECONV) wait_async0();
  __syncthreads();

  // ---- main loop: ping-pong pipeline, one barrier per k-step ----
  for (int ks = 0; ks < NSTEP; ++ks) {
    const int cur = ks & 1;
    const int nxt = cur ^ 1;

    // issue next-panel transfers first (overlap with the WMMAs below)
    if (ks + 1 < NSTEP) {
      if (PRECONV) {
        const char* wb = (const char*)w2bf + (ks + 1) * (BK * 2);
        const unsigned bb = (ks & 1) ? blsbase0 : blsbase1;
#pragma unroll
        for (int j = 0; j < 2; ++j)
          async_load_b128(bb + bloff[j], wb, bgoff[j]);
      }
      const float* xk = x + (ks + 1) * BK;
      const float* wk = w2 + (ks + 1) * BK;
#pragma unroll
      for (int i = 0; i < 4; ++i) {
        ra[i] = *(const float4*)(xk + aoff[i]);
        if (!PRECONV) rb[i] = *(const float4*)(wk + boff[i]);
      }
    }
    if (ks + 2 < NSTEP) {
      // global_prefetch_b8: warm L2 for the k+2 A panel
      __builtin_prefetch(x + (ks + 2) * BK + aoff[t & 3], 0, 1);
    }

    // fragment loads (ds_load_b128), ISA 16-bit A/B layout:
    // lane L: M(or N) = L%16; K halves [g*8..g*8+7] and [16+g*8..16+g*8+7]
    FragBF a[4], b[2];
#pragma unroll
    for (int ms = 0; ms < 4; ++ms) {
      const unsigned* p = &Als[cur][afrag[ms]];
      a[ms].q[0] = *(const uint4*)p;
      a[ms].q[1] = *(const uint4*)(p + 8);
    }
#pragma unroll
    for (int ns = 0; ns < 2; ++ns) {
      const unsigned* p = &Bls[cur][bfrag[ns]];
      b[ns].q[0] = *(const uint4*)p;
      b[ns].q[1] = *(const uint4*)(p + 8);
    }

#pragma unroll
    for (int ms = 0; ms < 4; ++ms)
#pragma unroll
      for (int ns = 0; ns < 2; ++ns)
        c[ms][ns] = __builtin_amdgcn_wmma_f32_16x16x32_bf16(
            false, a[ms].v, false, b[ns].v, (short)0, c[ms][ns], false, false);

    // convert + store next A panel behind the WMMAs
    if (ks + 1 < NSTEP) {
#pragma unroll
      for (int i = 0; i < 4; ++i) {
        uint2 ap; ap.x = pack_bf16x2(ra[i].x, ra[i].y); ap.y = pack_bf16x2(ra[i].z, ra[i].w);
        *(uint2*)&Als[nxt][lsoff[i]] = ap;
        if (!PRECONV) {
          uint2 bp; bp.x = pack_bf16x2(rb[i].x, rb[i].y); bp.y = pack_bf16x2(rb[i].z, rb[i].w);
          *(uint2*)&Bls[nxt][lsoff[i]] = bp;
        }
      }
    }
    if (PRECONV) wait_async0();
    __syncthreads();
  }

  // ---- epilogue: Kronecker w1 combine, fully in-lane over C VGPRs ----
  // C tile: VGPR r <-> subrow r of token (row_base/8 + g); lane%16 <-> q.
#pragma unroll
  for (int ms = 0; ms < 4; ++ms) {
    const int row_base = m0 + wm * 64 + ms * 16;
    const int tok = (row_base >> 3) + g;           // token id 0..16383
#pragma unroll
    for (int ns = 0; ns < 2; ++ns) {
      const int q = n0 + wn * 32 + ns * 16 + lm;
      float* op = out + (size_t)tok * 4096 + q;
      const v8f cc = c[ms][ns];
#pragma unroll
      for (int p = 0; p < 8; ++p) {
        float acc = 0.f;
#pragma unroll
        for (int r = 0; r < 8; ++r) acc += w1[p * 8 + r] * cc[r];  // scalar (SGPR) w1
        op[(size_t)p * 512] = acc * 0.25f;
      }
    }
  }
}

// One-shot f32 -> packed bf16 conversion of w2 (512x512) into workspace.
__global__ __launch_bounds__(256)
void w2_to_bf16_kernel(const float* __restrict__ w2, unsigned* __restrict__ w2bf) {
  const int i = blockIdx.x * 256 + threadIdx.x;       // 0..131071 packs
  const float2 v = *(const float2*)(w2 + i * 2);
  w2bf[i] = pack_bf16x2(v.x, v.y);
}

extern "C" void kernel_launch(void* const* d_in, const int* in_sizes, int n_in,
                              void* d_out, int out_size, void* d_ws, size_t ws_size,
                              hipStream_t stream) {
  const float* x  = (const float*)d_in[0];   // (4,4096,4096) f32
  const float* w1 = (const float*)d_in[1];   // (8,8) f32
  const float* w2 = (const float*)d_in[2];   // (512,512) f32
  float* out = (float*)d_out;                // (4,4096,4096) f32

  dim3 grid(ROWS / BM, NDIM / BN);           // 1024 x 4
  dim3 block(256);

  const size_t need = (size_t)KDIM * NDIM * 2;   // 512 KB bf16 copy of w2
  if (ws_size >= need) {
    unsigned short* w2bf = (unsigned short*)d_ws;
    w2_to_bf16_kernel<<<KDIM * NDIM / 2 / 256, 256, 0, stream>>>(w2, (unsigned*)w2bf);
    lokr_wmma_kernel<true><<<grid, block, 0, stream>>>(x, w1, w2, w2bf, out);
  } else {
    lokr_wmma_kernel<false><<<grid, block, 0, stream>>>(x, w1, w2, nullptr, out);
  }
}
